// Decoder_1700807049879
// MI455X (gfx1250) — matrix operand order repaired and checked by reading, not compiled, not run on hardware
//
#include <hip/hip_runtime.h>

typedef _Float16 h16;
typedef __attribute__((ext_vector_type(16))) _Float16 v16h;
typedef __attribute__((ext_vector_type(8)))  _Float16 v8h;
typedef __attribute__((ext_vector_type(4)))  _Float16 v4h;
typedef __attribute__((ext_vector_type(8)))  float    v8f;
typedef __attribute__((ext_vector_type(4)))  int      v4i;

// ---------------- generic 16B copier + optional async-to-LDS path ----------------

__device__ __forceinline__ void copy8(h16* dst, const h16* src) {
  *(v8h*)dst = *(const v8h*)src;
}

#if defined(__has_builtin)
#if __has_builtin(__builtin_amdgcn_global_load_async_to_lds_b128)
#define ASYNC_LDS 1
#endif
#endif

#ifdef ASYNC_LDS
typedef __attribute__((address_space(1))) v4i g_v4i;  // global (__device__) AS
typedef __attribute__((address_space(3))) v4i l_v4i;  // LDS AS
__device__ __forceinline__ void copy8_async(h16* dst, const h16* src) {
  __builtin_amdgcn_global_load_async_to_lds_b128((g_v4i*)src, (l_v4i*)dst, 0,
                                                 0);
}
__device__ __forceinline__ void async_wait() {
#if __has_builtin(__builtin_amdgcn_s_wait_asynccnt)
  __builtin_amdgcn_s_wait_asynccnt(0);
#else
  asm volatile("s_wait_asynccnt 0x0" ::: "memory");
#endif
}
#else
#define copy8_async copy8
__device__ __forceinline__ void async_wait() {}
#endif

// ---------------- WMMA helpers (wave32, 16x16x32 f16 -> f32) ----------------

__device__ __forceinline__ int lane_id() { return threadIdx.x & 31; }

// A fragment: 16x32 f16, row-major with stride lda (lda*2B must be 16B mult).
__device__ __forceinline__ v16h frag_a(const h16* s, int lda) {
  const int l = lane_id();
  const h16* p = s + (l & 15) * lda + ((l >> 4) << 3);
  v8h lo = *(const v8h*)p;
  v8h hi = *(const v8h*)(p + 16);
  return __builtin_shufflevector(lo, hi, 0, 1, 2, 3, 4, 5, 6, 7, 8, 9, 10, 11,
                                 12, 13, 14, 15);
}

// B fragment from TRANSPOSED storage Bt[n][k] (K contiguous, stride ldk).
// lane l -> col = l&15 ; K = (l>>4)*8 + [0..7] and 16 + (l>>4)*8 + [0..7].
__device__ __forceinline__ v16h frag_bt(const h16* bt, int ldk) {
  const int l = lane_id();
  const h16* p = bt + (l & 15) * ldk + ((l >> 4) << 3);
  v8h lo = *(const v8h*)p;
  v8h hi = *(const v8h*)(p + 16);
  return __builtin_shufflevector(lo, hi, 0, 1, 2, 3, 4, 5, 6, 7, 8, 9, 10, 11,
                                 12, 13, 14, 15);
}

__device__ __forceinline__ v8f wmma_f16(v16h a, v16h b, v8f c) {
  return __builtin_amdgcn_wmma_f32_16x16x32_f16(false, a, false, b, (short)0, c,
                                                false, false);
}

__device__ __forceinline__ void cidx(int& row0, int& col) {
  const int l = lane_id();
  col = l & 15;
  row0 = (l >> 4) << 3;
}

// ---------------- prep kernels ----------------

__global__ void k_prep_row(h16* crow, h16* srow) {
  int i = blockIdx.x * 256 + threadIdx.x;
  if (i >= 80 * 128) return;
  int kw = i / 128, w = i % 128;
  float c = 0.f, s = 0.f;
  if (kw < 65) {
    float ang = 6.283185307179586f * (float)(kw * w) / 128.f;
    const float inv = 0.08838834764831845f;  // 1/sqrt(128)
    c = __cosf(ang) * inv;
    s = -__sinf(ang) * inv;
  }
  crow[i] = (h16)c;
  srow[i] = (h16)s;
}

__global__ void k_prep_col(h16* cc, h16* sp) {
  int i = blockIdx.x * 256 + threadIdx.x;
  if (i >= 128 * 128) return;
  int r = i / 128, c = i % 128;
  float ang = 6.283185307179586f * (float)(r * c) / 128.f;
  const float inv = 0.08838834764831845f;
  cc[i] = (h16)(__cosf(ang) * inv);
  sp[i] = (h16)(__sinf(ang) * inv);
}

__global__ void k_prep_irow(h16* cir, h16* sir) {
  int i = blockIdx.x * 256 + threadIdx.x;
  if (i >= 128 * 96) return;
  int w = i / 96, kw = i % 96;
  float c = 0.f, s = 0.f;
  if (kw < 65) {
    float ck = (kw == 0 || kw == 64) ? 1.f : 2.f;
    float ang = 6.283185307179586f * (float)(kw * w) / 128.f;
    const float inv = 0.08838834764831845f;
    c = ck * __cosf(ang) * inv;
    s = -ck * __sinf(ang) * inv;
  }
  cir[i] = (h16)c;
  sir[i] = (h16)s;
}

// w1 [192][96] and w2 [96][192] are ALREADY in transposed-B layout Bt[n][k].
__global__ void k_cvt_h16(const float* __restrict__ src, h16* __restrict__ dst,
                          int n) {
  int i = blockIdx.x * 256 + threadIdx.x;
  if (i < n) dst[i] = (h16)src[i];
}

__global__ void k_zero(float* p, int n) {
  int i = blockIdx.x * 256 + threadIdx.x;
  if (i < n) p[i] = 0.f;
}

// ---------------- pwconv1 + SOA : [131072 x 96] x w1^T -> [x 192] ----------------

__global__ __launch_bounds__(256) void k_pwconv1(
    const float* __restrict__ x, const h16* __restrict__ w1t,
    const float* __restrict__ sc, const float* __restrict__ bi,
    h16* __restrict__ xpre) {
  __shared__ h16 As[64 * 104];   // 64 pixels x K=96 (pad 104)
  __shared__ h16 Bt[192 * 104];  // Bt[n=192][k=96] (pad 104)
  const int tid = threadIdx.x;
  const int p0 = blockIdx.x * 64;
  // A fill: f32 -> f16, 4 elements per 16B global load
  for (int i = tid * 4; i < 64 * 96; i += 1024) {
    const float4 v = *(const float4*)(x + (size_t)p0 * 96 + i);
    v4h hv = {(h16)v.x, (h16)v.y, (h16)v.z, (h16)v.w};
    int r = i / 96, c = i % 96;
    *(v4h*)(As + r * 104 + c) = hv;
  }
  // prefetch next pixel tile
  if (tid < 64)
    __builtin_prefetch(x + (size_t)(p0 + 64 + tid) * 96, 0, 3);
  // B fill: rows n=192, K=96
  for (int i = tid; i < 192 * 12; i += 256) {
    int r = i / 12, c8 = (i % 12) * 8;
    copy8(Bt + r * 104 + c8, w1t + r * 96 + c8);
  }
  __syncthreads();
  const float s = sc[0], b = bi[0];
  int row0, col;
  cidx(row0, col);
  const int wave = tid >> 5;
  for (int job = wave; job < 48; job += 8) {
    const int mt = job & 3, nt = job >> 2;
    v8f acc = {};
#pragma unroll
    for (int kk = 0; kk < 3; ++kk) {
      v16h a = frag_a(As + mt * 16 * 104 + kk * 32, 104);
      v16h bb = frag_bt(Bt + nt * 16 * 104 + kk * 32, 104);
      acc = wmma_f16(a, bb, acc);
    }
#pragma unroll
    for (int r = 0; r < 8; ++r) {
      float v = acc[r];
      float rv = v > 0.f ? v : 0.f;
      xpre[(size_t)(p0 + mt * 16 + row0 + r) * 192 + nt * 16 + col] =
          (h16)(s * rv * rv + b);
    }
  }
}

// ---------------- row rFFT as GEMM: per (b,h): [80x128] x [128x48] ----------------

__global__ __launch_bounds__(256) void k_rowdft(
    const h16* __restrict__ xpre, const h16* __restrict__ crow,
    const h16* __restrict__ srow, h16* __restrict__ outRe,
    h16* __restrict__ outIm) {
  __shared__ h16 Ac[80 * 136];
  __shared__ h16 Asn[80 * 136];
  __shared__ h16 Bt[48 * 136];  // Bt[n=48][k=128]
  const int tid = threadIdx.x;
  const int slab = blockIdx.x;  // b*128+h
  const int nsl = blockIdx.y;   // 0..3
  for (int i = tid; i < 80 * 16; i += 256) {
    int r = i / 16, c8 = (i % 16) * 8;
    copy8(Ac + r * 136 + c8, crow + r * 128 + c8);
    copy8(Asn + r * 136 + c8, srow + r * 128 + c8);
  }
  const h16* src = xpre + (size_t)slab * (128 * 192) + nsl * 48;
  for (int i = tid; i < 128 * 6; i += 256) {
    int r = i / 6, c8 = (i % 6) * 8;
    v8h v = *(const v8h*)(src + (size_t)r * 192 + c8);
#pragma unroll
    for (int j = 0; j < 8; ++j) Bt[(c8 + j) * 136 + r] = v[j];
  }
  __syncthreads();
  int row0, col;
  cidx(row0, col);
  const int wave = tid >> 5;
  for (int job = wave; job < 30; job += 8) {
    const int o = job / 15;  // 0 = re, 1 = im
    const int p = job % 15;
    const int mt = p % 5, nt = p / 5;
    const h16* A = o ? Asn : Ac;
    v8f acc = {};
#pragma unroll
    for (int kk = 0; kk < 4; ++kk) {
      v16h a = frag_a(A + mt * 16 * 136 + kk * 32, 136);
      v16h bb = frag_bt(Bt + nt * 16 * 136 + kk * 32, 136);
      acc = wmma_f16(a, bb, acc);
    }
    h16* out = o ? outIm : outRe;
#pragma unroll
    for (int r = 0; r < 8; ++r) {
      int kw = mt * 16 + row0 + r;
      if (kw < 65)
        out[((size_t)slab * 65 + kw) * 192 + nsl * 48 + nt * 16 + col] =
            (h16)acc[r];
    }
  }
}

// ------- column DFT (complex, shared fwd/inv): per (b,kw) slab -------
// fwd (e^-i): re = Cc*re + Sp*im (sB=+1), im = Cc*im - Sp*re (sD=-1)
// inv (e^+i): re = Cc*re - Sp*im (sB=-1), im = Cc*im + Sp*re (sD=+1)

__global__ __launch_bounds__(256) void k_coldft(
    const h16* __restrict__ inRe, const h16* __restrict__ inIm,
    const h16* __restrict__ ccG, const h16* __restrict__ spG,
    h16* __restrict__ outRe, h16* __restrict__ outIm, int inBStride,
    int inKwStride, int inRowStride, float sB, float sD) {
  __shared__ h16 Ac[64 * 136];
  __shared__ h16 Asn[64 * 136];
  __shared__ h16 BtRe[48 * 136];
  __shared__ h16 BtIm[48 * 136];
  const int tid = threadIdx.x;
  const int slab = blockIdx.x;  // b*65+kw
  const int b = slab / 65, kw = slab % 65;
  const int khh = blockIdx.y;  // output row half
  const int nsl = blockIdx.z;  // 0..3
  // A fill via async global->LDS when available
  for (int i = tid; i < 64 * 16; i += 256) {
    int r = i / 16, c8 = (i % 16) * 8;
    copy8_async(Ac + r * 136 + c8, ccG + khh * 8192 + r * 128 + c8);
    copy8_async(Asn + r * 136 + c8, spG + khh * 8192 + r * 128 + c8);
  }
  const size_t ib = (size_t)b * inBStride + (size_t)kw * inKwStride + nsl * 48;
  for (int i = tid; i < 128 * 6; i += 256) {
    int r = i / 6, c8 = (i % 6) * 8;
    v8h vr = *(const v8h*)(inRe + ib + (size_t)r * inRowStride + c8);
    v8h vi = *(const v8h*)(inIm + ib + (size_t)r * inRowStride + c8);
#pragma unroll
    for (int j = 0; j < 8; ++j) {
      BtRe[(c8 + j) * 136 + r] = vr[j];
      BtIm[(c8 + j) * 136 + r] = vi[j];
    }
  }
  async_wait();
  __syncthreads();
  int row0, col;
  cidx(row0, col);
  const int wave = tid >> 5;
  const size_t ob = ((size_t)slab * 128 + khh * 64) * 192 + nsl * 48;
  for (int job = wave; job < 12; job += 8) {
    const int mt = job % 4, nt = job / 4;
    v8f aA = {}, aB = {}, aC = {}, aD = {};
#pragma unroll
    for (int kk = 0; kk < 4; ++kk) {
      v16h ac = frag_a(Ac + mt * 16 * 136 + kk * 32, 136);
      v16h as = frag_a(Asn + mt * 16 * 136 + kk * 32, 136);
      v16h br = frag_bt(BtRe + nt * 16 * 136 + kk * 32, 136);
      v16h bim = frag_bt(BtIm + nt * 16 * 136 + kk * 32, 136);
      aA = wmma_f16(ac, br, aA);
      aB = wmma_f16(as, bim, aB);
      aC = wmma_f16(ac, bim, aC);
      aD = wmma_f16(as, br, aD);
    }
#pragma unroll
    for (int r = 0; r < 8; ++r) {
      int ro = mt * 16 + row0 + r;
      outRe[ob + (size_t)ro * 192 + nt * 16 + col] = (h16)(aA[r] + sB * aB[r]);
      outIm[ob + (size_t)ro * 192 + nt * 16 + col] = (h16)(aC[r] + sD * aD[r]);
    }
  }
}

// ---------------- routed spectral multiply (in-place, elementwise) ----------------

__global__ __launch_bounds__(256) void k_specmul(
    h16* __restrict__ xfre, h16* __restrict__ xfim,
    const h16* __restrict__ fre, const h16* __restrict__ fim,
    const float* __restrict__ rw) {
  size_t idx = (size_t)blockIdx.x * 256 + threadIdx.x;
  const size_t total = (size_t)8 * 65 * 128 * 192;
  if (idx >= total) return;
  int m = (int)(idx % 192);
  size_t t = idx / 192;
  int kh = (int)(t % 128);
  size_t t2 = t / 128;
  int kw = (int)(t2 % 65);
  int b = (int)(t2 / 65);
  size_t f0 = ((size_t)kw * 128 + kh) * 192 + m;
  const size_t fplane = (size_t)65 * 128 * 192;
  float cre = 0.f, cim = 0.f;
#pragma unroll
  for (int s = 0; s < 3; ++s) {
    float rs = rw[((size_t)b * 3 + s) * 192 + m];
    cre += (float)fre[s * fplane + f0] * rs;
    cim += (float)fim[s * fplane + f0] * rs;
  }
  float a = (float)xfre[idx], c = (float)xfim[idx];
  xfre[idx] = (h16)(a * cre - c * cim);
  xfim[idx] = (h16)(a * cim + c * cre);
}

// ---------------- inverse row rFFT (Hermitian folded): per (b,h) ----------------

__global__ __launch_bounds__(256) void k_irowdft(
    const h16* __restrict__ xiRe, const h16* __restrict__ xiIm,
    const h16* __restrict__ cir, const h16* __restrict__ sir,
    h16* __restrict__ xsp) {
  __shared__ h16 Ac[64 * 104];
  __shared__ h16 Asn[64 * 104];
  __shared__ h16 BtRe[48 * 104];
  __shared__ h16 BtIm[48 * 104];
  const int tid = threadIdx.x;
  const int slab = blockIdx.x;  // b*128+h
  const int b = slab >> 7, h = slab & 127;
  const int wh = blockIdx.y;   // 0/1 w-half
  const int nsl = blockIdx.z;  // 0..3
  for (int i = tid; i < 64 * 12; i += 256) {
    int r = i / 12, c8 = (i % 12) * 8;
    copy8(Ac + r * 104 + c8, cir + wh * 64 * 96 + r * 96 + c8);
    copy8(Asn + r * 104 + c8, sir + wh * 64 * 96 + r * 96 + c8);
  }
  const size_t src0 = ((size_t)b * 65 * 128 + h) * 192 + nsl * 48;
  for (int i = tid; i < 96 * 6; i += 256) {
    int r = i / 6, c8 = (i % 6) * 8;
    v8h vr, vi;
    if (r < 65) {
      vr = *(const v8h*)(xiRe + src0 + (size_t)r * (128 * 192) + c8);
      vi = *(const v8h*)(xiIm + src0 + (size_t)r * (128 * 192) + c8);
    } else {
      vr = (v8h)(h16)0.f;
      vi = (v8h)(h16)0.f;
    }
#pragma unroll
    for (int j = 0; j < 8; ++j) {
      BtRe[(c8 + j) * 104 + r] = vr[j];
      BtIm[(c8 + j) * 104 + r] = vi[j];
    }
  }
  __syncthreads();
  int row0, col;
  cidx(row0, col);
  const int wave = tid >> 5;
  for (int job = wave; job < 12; job += 8) {
    const int mt = job % 4, nt = job / 4;
    v8f acc = {};
#pragma unroll
    for (int kk = 0; kk < 3; ++kk) {
      v16h a = frag_a(Ac + mt * 16 * 104 + kk * 32, 104);
      v16h bb = frag_bt(BtRe + nt * 16 * 104 + kk * 32, 104);
      acc = wmma_f16(a, bb, acc);
    }
#pragma unroll
    for (int kk = 0; kk < 3; ++kk) {
      v16h a = frag_a(Asn + mt * 16 * 104 + kk * 32, 104);
      v16h bb = frag_bt(BtIm + nt * 16 * 104 + kk * 32, 104);
      acc = wmma_f16(a, bb, acc);
    }
#pragma unroll
    for (int r = 0; r < 8; ++r) {
      int w = wh * 64 + mt * 16 + row0 + r;
      xsp[(size_t)slab * (128 * 192) + (size_t)w * 192 + nsl * 48 + nt * 16 +
          col] = (h16)acc[r];
    }
  }
}

// ---------------- pwconv2: [131072 x 192] x w2^T -> f32 out ----------------

__global__ __launch_bounds__(256) void k_pwconv2(const h16* __restrict__ xsp,
                                                 const h16* __restrict__ w2t,
                                                 float* __restrict__ out) {
  __shared__ h16 As[64 * 200];  // 64 pixels x K=192 (pad 200)
  __shared__ h16 Bt[96 * 200];  // Bt[n=96][k=192]
  const int tid = threadIdx.x;
  const int p0 = blockIdx.x * 64;
  for (int i = tid; i < 64 * 24; i += 256) {
    int r = i / 24, c8 = (i % 24) * 8;
    copy8(As + r * 200 + c8, xsp + (size_t)(p0 + r) * 192 + c8);
  }
  for (int i = tid; i < 96 * 24; i += 256) {
    int r = i / 24, c8 = (i % 24) * 8;
    copy8(Bt + r * 200 + c8, w2t + r * 192 + c8);
  }
  __syncthreads();
  int row0, col;
  cidx(row0, col);
  const int wave = tid >> 5;
  for (int job = wave; job < 24; job += 8) {
    const int mt = job & 3, nt = job >> 2;
    v8f acc = {};
#pragma unroll
    for (int kk = 0; kk < 6; ++kk) {
      v16h a = frag_a(As + mt * 16 * 200 + kk * 32, 200);
      v16h bb = frag_bt(Bt + nt * 16 * 200 + kk * 32, 200);
      acc = wmma_f16(a, bb, acc);
    }
#pragma unroll
    for (int r = 0; r < 8; ++r)
      out[(size_t)(p0 + mt * 16 + row0 + r) * 96 + nt * 16 + col] = acc[r];
  }
}

// ---------------- routing branch ----------------

__global__ __launch_bounds__(256) void k_gctx(const float* __restrict__ x,
                                              float* __restrict__ gctx) {
  __shared__ float red[256];
  int b = blockIdx.x / 96, c = blockIdx.x % 96;
  float s = 0.f;
  const float* xb = x + (size_t)b * 16384 * 96 + c;
  for (int p = threadIdx.x; p < 16384; p += 256) s += xb[(size_t)p * 96];
  red[threadIdx.x] = s;
  __syncthreads();
  for (int st = 128; st > 0; st >>= 1) {
    if (threadIdx.x < st) red[threadIdx.x] += red[threadIdx.x + st];
    __syncthreads();
  }
  if (threadIdx.x == 0) gctx[blockIdx.x] = red[0] * (1.f / 16384.f);
}

__global__ __launch_bounds__(256) void k_sctx(
    const float* __restrict__ x, const float* __restrict__ spw,
    const float* __restrict__ gam, const float* __restrict__ bet,
    const float* __restrict__ mean, const float* __restrict__ var,
    float* __restrict__ sctxa) {
  __shared__ float sh[48];
  const int b = blockIdx.x;
  const int p = blockIdx.y * 256 + threadIdx.x;
  if (threadIdx.x < 48) sh[threadIdx.x] = 0.f;
  __syncthreads();
  float acc[48];
#pragma unroll
  for (int j = 0; j < 48; ++j) acc[j] = 0.f;
  const float* xp = x + ((size_t)b * 16384 + p) * 96;
  for (int c0 = 0; c0 < 96; c0 += 16) {
    float xv[16];
#pragma unroll
    for (int cc = 0; cc < 16; ++cc) xv[cc] = xp[c0 + cc];
#pragma unroll
    for (int j = 0; j < 48; ++j) {
      float d = 0.f;
#pragma unroll
      for (int cc = 0; cc < 16; ++cc) d += xv[cc] * spw[j * 96 + c0 + cc];
      acc[j] += d;
    }
  }
#pragma unroll
  for (int j = 0; j < 48; ++j) {
    float v = (acc[j] - mean[j]) * rsqrtf(var[j] + 1e-5f) * gam[j] + bet[j];
    v = v > 0.f ? v : 0.f;
    atomicAdd(&sh[j], v);
  }
  __syncthreads();
  if (threadIdx.x < 48) atomicAdd(&sctxa[b * 48 + threadIdx.x], sh[threadIdx.x]);
}

__global__ __launch_bounds__(256) void k_mlp(
    const float* __restrict__ gctx, const float* __restrict__ sctxa,
    const float* __restrict__ fc1, const float* __restrict__ msc,
    const float* __restrict__ mbi, const float* __restrict__ fc2,
    float* __restrict__ rw) {
  __shared__ float fused[8 * 144];
  __shared__ float hm[8 * 36];
  __shared__ float lg[8 * 576];
  const int tid = threadIdx.x;
  for (int i = tid; i < 8 * 144; i += 256) {
    int b = i / 144, j = i % 144;
    fused[i] = (j < 96) ? gctx[b * 96 + j]
                        : sctxa[b * 48 + (j - 96)] * (1.f / 16384.f);
  }
  __syncthreads();
  if (tid < 8 * 36) {
    int b = tid / 36, j = tid % 36;
    float d = 0.f;
    for (int k = 0; k < 144; ++k) d += fc1[j * 144 + k] * fused[b * 144 + k];
    float rv = d > 0.f ? d : 0.f;
    hm[tid] = msc[0] * rv * rv + mbi[0];
  }
  __syncthreads();
  for (int i = tid; i < 8 * 576; i += 256) {
    int b = i / 576, o = i % 576;
    float d = 0.f;
    for (int k = 0; k < 36; ++k) d += fc2[o * 36 + k] * hm[b * 36 + k];
    lg[i] = d;
  }
  __syncthreads();
  for (int i = tid; i < 8 * 192; i += 256) {
    int b = i / 192, m = i % 192;
    float v0 = lg[b * 576 + m], v1 = lg[b * 576 + 192 + m],
          v2 = lg[b * 576 + 384 + m];
    float mx = fmaxf(v0, fmaxf(v1, v2));
    float e0 = __expf(v0 - mx), e1 = __expf(v1 - mx), e2 = __expf(v2 - mx);
    float inv = 1.f / (e0 + e1 + e2);
    rw[(b * 3 + 0) * 192 + m] = e0 * inv;
    rw[(b * 3 + 1) * 192 + m] = e1 * inv;
    rw[(b * 3 + 2) * 192 + m] = e2 * inv;
  }
}

// ---------------- bicubic filter resize (align_corners, a=-0.75) ----------------

__device__ __forceinline__ float cubicw(float t) {
  const float a = -0.75f;
  t = fabsf(t);
  if (t <= 1.f) return ((a + 2.f) * t - (a + 3.f)) * t * t + 1.f;
  if (t < 2.f) return (((t - 5.f) * t + 8.f) * t - 4.f) * a;
  return 0.f;
}

__global__ __launch_bounds__(192) void k_resize(
    const float* __restrict__ cw0, const float* __restrict__ cw1,
    const float* __restrict__ cw2, h16* __restrict__ fre,
    h16* __restrict__ fim) {
  const int bi = blockIdx.x;  // s*65*128 + kw*128 + kh
  const int s = bi / (65 * 128);
  const int rem = bi % (65 * 128);
  const int kw = rem / 128;
  const int kh = rem % 128;
  const int m = threadIdx.x;
  int Hs, Ws;
  const float* cw;
  if (s == 0) { Hs = 16; Ws = 9; cw = cw0; }
  else if (s == 1) { Hs = 8; Ws = 4; cw = cw1; }
  else { Hs = 24; Ws = 13; cw = cw2; }
  float shc = (float)kh * (float)(Hs - 1) / 127.f;
  float swc = (float)kw * (float)(Ws - 1) / 64.f;
  int fh = (int)floorf(shc);
  int fw = (int)floorf(swc);
  float re = 0.f, im = 0.f;
  for (int ta = -1; ta <= 2; ++ta) {
    float whv = cubicw(shc - (float)(fh + ta));
    int ih = fh + ta;
    ih = ih < 0 ? 0 : (ih > Hs - 1 ? Hs - 1 : ih);
    for (int tb = -1; tb <= 2; ++tb) {
      float wwv = cubicw(swc - (float)(fw + tb));
      int iw = fw + tb;
      iw = iw < 0 ? 0 : (iw > Ws - 1 ? Ws - 1 : iw);
      const float* p = cw + (((size_t)ih * Ws + iw) * 192 + m) * 2;
      re += whv * wwv * p[0];
      im += whv * wwv * p[1];
    }
  }
  size_t o = (((size_t)s * 65 + kw) * 128 + kh) * 192 + m;
  fre[o] = (h16)re;
  fim[o] = (h16)im;
}

// ---------------- host launch ----------------

extern "C" void kernel_launch(void* const* d_in, const int* in_sizes, int n_in,
                              void* d_out, int out_size, void* d_ws,
                              size_t ws_size, hipStream_t stream) {
  (void)in_sizes; (void)n_in; (void)out_size; (void)ws_size;
  const float* x    = (const float*)d_in[0];
  const float* w1   = (const float*)d_in[1];
  const float* soas = (const float*)d_in[2];
  const float* soab = (const float*)d_in[3];
  const float* cw0  = (const float*)d_in[4];
  const float* cw1  = (const float*)d_in[5];
  const float* cw2  = (const float*)d_in[6];
  const float* spw  = (const float*)d_in[7];
  const float* gam  = (const float*)d_in[8];
  const float* bet  = (const float*)d_in[9];
  const float* mean = (const float*)d_in[10];
  const float* var  = (const float*)d_in[11];
  const float* fc1  = (const float*)d_in[12];
  const float* msc  = (const float*)d_in[13];
  const float* mbi  = (const float*)d_in[14];
  const float* fc2  = (const float*)d_in[15];
  const float* w2   = (const float*)d_in[16];
  float* out = (float*)d_out;

  char* ws = (char*)d_ws;
  size_t off = 0;
  auto alloc = [&](size_t bytes) {
    void* p = ws + off;
    off = (off + bytes + 255) & ~(size_t)255;
    return p;
  };
  const size_t NXPRE = (size_t)25165824;  // 8*128*128*192
  const size_t NSPEC = (size_t)12779520;  // 8*128*65*192
  const size_t NFILT = (size_t)4792320;   // 3*65*128*192
  h16* XPRE  = (h16*)alloc(NXPRE * 2);    // also reused as XSP
  h16* XF1RE = (h16*)alloc(NSPEC * 2);    // also reused as XI1RE
  h16* XF1IM = (h16*)alloc(NSPEC * 2);    // also reused as XI1IM
  h16* XFRE  = (h16*)alloc(NSPEC * 2);
  h16* XFIM  = (h16*)alloc(NSPEC * 2);
  h16* FRE   = (h16*)alloc(NFILT * 2);
  h16* FIM   = (h16*)alloc(NFILT * 2);
  h16* CROW  = (h16*)alloc(10240 * 2);
  h16* SROW  = (h16*)alloc(10240 * 2);
  h16* CC    = (h16*)alloc(16384 * 2);
  h16* SP    = (h16*)alloc(16384 * 2);
  h16* CIR   = (h16*)alloc(12288 * 2);
  h16* SIR   = (h16*)alloc(12288 * 2);
  h16* W1T   = (h16*)alloc(18432 * 2);
  h16* W2T   = (h16*)alloc(18432 * 2);
  float* GCTX  = (float*)alloc(768 * 4);
  float* SCTXA = (float*)alloc(384 * 4);
  float* RW    = (float*)alloc(4608 * 4);

  k_prep_row<<<40, 256, 0, stream>>>(CROW, SROW);
  k_prep_col<<<64, 256, 0, stream>>>(CC, SP);
  k_prep_irow<<<48, 256, 0, stream>>>(CIR, SIR);
  k_cvt_h16<<<72, 256, 0, stream>>>(w1, W1T, 18432);
  k_cvt_h16<<<72, 256, 0, stream>>>(w2, W2T, 18432);
  k_zero<<<2, 256, 0, stream>>>(SCTXA, 384);

  k_pwconv1<<<2048, 256, 0, stream>>>(x, W1T, soas, soab, XPRE);
  k_gctx<<<768, 256, 0, stream>>>(x, GCTX);
  k_sctx<<<dim3(8, 64), 256, 0, stream>>>(x, spw, gam, bet, mean, var, SCTXA);
  k_mlp<<<1, 256, 0, stream>>>(GCTX, SCTXA, fc1, msc, mbi, fc2, RW);
  k_resize<<<3 * 65 * 128, 192, 0, stream>>>(cw0, cw1, cw2, FRE, FIM);

  k_rowdft<<<dim3(1024, 4), 256, 0, stream>>>(XPRE, CROW, SROW, XF1RE, XF1IM);
  // forward column FFT: input XF1 [b][h][65][192]
  k_coldft<<<dim3(520, 2, 4), 256, 0, stream>>>(
      XF1RE, XF1IM, CC, SP, XFRE, XFIM, 128 * 65 * 192, 192, 65 * 192, 1.f,
      -1.f);
  k_specmul<<<49920, 256, 0, stream>>>(XFRE, XFIM, FRE, FIM, RW);
  // inverse column FFT: input XF [b][kw][kh][192]
  k_coldft<<<dim3(520, 2, 4), 256, 0, stream>>>(
      XFRE, XFIM, CC, SP, XF1RE, XF1IM, 65 * 128 * 192, 128 * 192, 192, -1.f,
      1.f);
  k_irowdft<<<dim3(1024, 2, 4), 256, 0, stream>>>(XF1RE, XF1IM, CIR, SIR,
                                                  XPRE);
  k_pwconv2<<<2048, 256, 0, stream>>>(XPRE, W2T, out);
}
